// OuterProduct_53944789238366
// MI455X (gfx1250) — matrix operand order repaired and checked by reading, not compiled, not run on hardware
//
#include <hip/hip_runtime.h>
#include <stdint.h>

#define N_NODE 50000
#define D_ATOM 512
#define D_PAIR 128
#define D_HID  32
#define N_EDGE 400000

typedef __attribute__((ext_vector_type(16))) __bf16 v16bf;
typedef __attribute__((ext_vector_type(8)))  float  v8f;
typedef __attribute__((ext_vector_type(8)))  unsigned int u32x8;

static __device__ __forceinline__ unsigned short f2bf(float f) {
  unsigned u = __builtin_bit_cast(unsigned, f);
  u += 0x7FFFu + ((u >> 16) & 1u);           // round-to-nearest-even
  return (unsigned short)(u >> 16);
}

static __device__ __forceinline__ unsigned pk_mul_bf16(unsigned a, unsigned b) {
  unsigned d;
  asm("v_pk_mul_bf16 %0, %1, %2" : "=v"(d) : "v"(a), "v"(b));
  return d;
}

static __device__ __forceinline__ v8f wmma_bf16(u32x8 a, u32x8 b, v8f c) {
  return __builtin_amdgcn_wmma_f32_16x16x32_bf16(
      false, __builtin_bit_cast(v16bf, a),
      false, __builtin_bit_cast(v16bf, b),
      (short)0, c, false, false);
}

// ---------------------------------------------------------------------------
// Kernel 0: permute W_out (128 x 1024 f32) -> bf16 B-fragments (contiguous k per lane half, matching the A-frags).
// Layout (u32 words): wb[ (((pt*32+h)*2 + c)*32 + lane)*4 + w ]
//   v = c*4+w (VGPR index 0..7), n = lane&15, k = (lane&16?16:0) + 2*v + {0,1}
//   value = W_out[pt*16+n][h*32+k]
// 65536 u32 total = 256 KB.
// ---------------------------------------------------------------------------
__global__ void prep_wb_kernel(const float* __restrict__ Wout,
                               unsigned* __restrict__ wb) {
  unsigned u = blockIdx.x * blockDim.x + threadIdx.x;   // 0..65535
  unsigned w    =  u        & 3u;
  unsigned lane = (u >> 2)  & 31u;
  unsigned c    = (u >> 7)  & 1u;
  unsigned h    = (u >> 8)  & 31u;
  unsigned pt   =  u >> 13;
  unsigned v = c * 4u + w;
  unsigned n = lane & 15u;
  unsigned k = ((lane & 16u) ? 16u : 0u) + 2u * v;
  const float* src = Wout + (size_t)(pt * 16u + n) * 1024u + h * 32u + k;
  unsigned lo = f2bf(src[0]);
  unsigned hi = f2bf(src[1]);
  wb[u] = lo | (hi << 16);
}

// ---------------------------------------------------------------------------
// Kernel 1: ab = m @ W_in^T + b_in  (f32 compute, store bf16 halves a/b).
// Block = 256 threads = 16 rows x 16 j-quads; m rows staged in LDS.
// ---------------------------------------------------------------------------
__global__ __launch_bounds__(256) void phase1_kernel(
    const float* __restrict__ m, const float* __restrict__ Win,
    const float* __restrict__ bin,
    unsigned short* __restrict__ aB, unsigned short* __restrict__ bB) {
  __shared__ float sm[16 * 512];
  const int tid = threadIdx.x;
  const int n0  = blockIdx.x * 16;

  const float4* m4 = (const float4*)(m + (size_t)n0 * 512);
  float4* s4 = (float4*)sm;
#pragma unroll
  for (int i = 0; i < 8; ++i) s4[tid + i * 256] = m4[tid + i * 256];
  __syncthreads();

  const int rt = tid >> 4;
  const int j0 = (tid & 15) * 4;
  const float4* W4   = (const float4*)Win;
  const float4* mrow = (const float4*)(sm + rt * 512);

  float a0 = 0.f, a1 = 0.f, a2 = 0.f, a3 = 0.f;
#pragma unroll 4
  for (int d4 = 0; d4 < 128; ++d4) {
    float4 mv = mrow[d4];
    float4 w0 = W4[(j0 + 0) * 128 + d4];
    float4 w1 = W4[(j0 + 1) * 128 + d4];
    float4 w2 = W4[(j0 + 2) * 128 + d4];
    float4 w3 = W4[(j0 + 3) * 128 + d4];
    a0 = fmaf(mv.x, w0.x, fmaf(mv.y, w0.y, fmaf(mv.z, w0.z, fmaf(mv.w, w0.w, a0))));
    a1 = fmaf(mv.x, w1.x, fmaf(mv.y, w1.y, fmaf(mv.z, w1.z, fmaf(mv.w, w1.w, a1))));
    a2 = fmaf(mv.x, w2.x, fmaf(mv.y, w2.y, fmaf(mv.z, w2.z, fmaf(mv.w, w2.w, a2))));
    a3 = fmaf(mv.x, w3.x, fmaf(mv.y, w3.y, fmaf(mv.z, w3.z, fmaf(mv.w, w3.w, a3))));
  }
  a0 += bin[j0 + 0]; a1 += bin[j0 + 1]; a2 += bin[j0 + 2]; a3 += bin[j0 + 3];

  unsigned p0 = (unsigned)f2bf(a0) | ((unsigned)f2bf(a1) << 16);
  unsigned p1 = (unsigned)f2bf(a2) | ((unsigned)f2bf(a3) << 16);
  const int n = n0 + rt;
  unsigned short* dst;
  int col;
  if (j0 < 32) { dst = aB; col = j0; } else { dst = bB; col = j0 - 32; }
  uint2* o = (uint2*)(dst + (size_t)n * 32 + col);
  *o = make_uint2(p0, p1);
}

// ---------------------------------------------------------------------------
// Kernel 2: z[e,p] = sum_{h,k} ae[e,h] be[e,k] W3[p,h,k] + b_out[p]
// Per wave: 2 edge tiles (32 edges) x all 128 pairs.
//   A-frag (be, WMMA A-layout) resident; per h: broadcast ae scale via
//   v_pk_mul_bf16, then 8 pt x 2 et WMMAs with B-frags from LDS.
// LDS: 256 KB Wb stage + 4 KB/wave ae-dup = 294912 B dynamic.
// ---------------------------------------------------------------------------
__global__ __launch_bounds__(256, 2) void pair_outer_kernel(
    const int* __restrict__ eidx,
    const unsigned short* __restrict__ aB,
    const unsigned short* __restrict__ bB,
    const unsigned* __restrict__ wbG,
    const float* __restrict__ bout,
    float* __restrict__ z) {
  extern __shared__ unsigned sh[];     // [0,65536): Wb ; then 1024 u32 per wave
  const int tid = threadIdx.x;

  { // stage Wb into LDS (coalesced b128)
    uint4* d = (uint4*)sh;
    const uint4* s = (const uint4*)wbG;
#pragma unroll
    for (int i = 0; i < 64; ++i) d[tid + i * 256] = s[tid + i * 256];
  }
  __syncthreads();

  const int wave = tid >> 5, lane = tid & 31;
  unsigned* aDup = sh + 65536 + wave * 1024;   // [et][h][m] u32 (bf16 dup'd)
  const int nlo    = lane & 15;
  const int laneHi = lane >> 4;                // 0 or 1
  const int aoff   = laneHi * 32;              // A-frag K-half byte offset

  float bias[8];
#pragma unroll
  for (int pt = 0; pt < 8; ++pt) bias[pt] = bout[pt * 16 + nlo];

  const int tpEnd = N_EDGE / 32;               // 12500 tile-pairs
  for (int tp = blockIdx.x * 8 + wave; tp < tpEnd; tp += gridDim.x * 8) {
    const int e0 = tp * 32;

    // ---- stage ae (dup'd bf16 pairs) : lane covers (et=laneHi, m=nlo) ----
    {
      const int eA   = e0 + laneHi * 16 + nlo;
      const int rowA = eidx[2 * eA + 0];
      const uint4* ar = (const uint4*)(aB + (size_t)rowA * 32);
      unsigned* dst = aDup + laneHi * 512;
#pragma unroll
      for (int q = 0; q < 4; ++q) {
        uint4 vq = ar[q];
        unsigned ws[4] = {vq.x, vq.y, vq.z, vq.w};
#pragma unroll
        for (int w = 0; w < 4; ++w) {
          int h = q * 8 + w * 2;
          unsigned lo = ws[w] & 0xFFFFu, hi = ws[w] >> 16;
          dst[(h + 0) * 16 + nlo] = lo | (lo << 16);
          dst[(h + 1) * 16 + nlo] = hi | (hi << 16);
        }
      }
    }

    // ---- gather be rows into WMMA A-layout fragments (both edge tiles) ----
    const int rB0 = eidx[2 * (e0 + nlo) + 1];
    const int rB1 = eidx[2 * (e0 + 16 + nlo) + 1];
    const char* pb0 = (const char*)bB + (size_t)rB0 * 64 + aoff;
    const char* pb1 = (const char*)bB + (size_t)rB1 * 64 + aoff;
    uint4 b0l = *(const uint4*)pb0, b0h = *(const uint4*)(pb0 + 16);
    uint4 b1l = *(const uint4*)pb1, b1h = *(const uint4*)(pb1 + 16);
    u32x8 be0 = {b0l.x, b0l.y, b0l.z, b0l.w, b0h.x, b0h.y, b0h.z, b0h.w};
    u32x8 be1 = {b1l.x, b1l.y, b1l.z, b1l.w, b1h.x, b1h.y, b1h.z, b1h.w};

    // ---- accumulators initialized with bias ----
    v8f acc[2][8];
#pragma unroll
    for (int et = 0; et < 2; ++et)
#pragma unroll
      for (int pt = 0; pt < 8; ++pt) {
        v8f a;
#pragma unroll
        for (int i = 0; i < 8; ++i) a[i] = bias[pt];
        acc[et][pt] = a;
      }

    // ---- main K loop over h ----
#pragma unroll 4
    for (int h = 0; h < 32; ++h) {
      unsigned s0 = aDup[h * 16 + nlo];
      unsigned s1 = aDup[512 + h * 16 + nlo];
      u32x8 A0, A1;
#pragma unroll
      for (int j = 0; j < 8; ++j) {
        A0[j] = pk_mul_bf16(s0, be0[j]);
        A1[j] = pk_mul_bf16(s1, be1[j]);
      }
#pragma unroll
      for (int pt = 0; pt < 8; ++pt) {
        const unsigned* bp = sh + (((pt * 32 + h) * 2) * 32 + lane) * 4;
        uint4 q0 = *(const uint4*)bp;
        uint4 q1 = *(const uint4*)(bp + 128);
        u32x8 B = {q0.x, q0.y, q0.z, q0.w, q1.x, q1.y, q1.z, q1.w};
        acc[0][pt] = wmma_bf16(A0, B, acc[0][pt]);
        acc[1][pt] = wmma_bf16(A1, B, acc[1][pt]);
      }
    }

    // ---- store (nontemporal, streaming output) ----
#pragma unroll
    for (int et = 0; et < 2; ++et)
#pragma unroll
      for (int pt = 0; pt < 8; ++pt) {
        v8f a = acc[et][pt];
#pragma unroll
        for (int r = 0; r < 8; ++r) {
          int row = e0 + et * 16 + r + 8 * laneHi;
          __builtin_nontemporal_store(a[r],
              z + (size_t)row * 128 + pt * 16 + nlo);
        }
      }
  }
}

// ---------------------------------------------------------------------------
extern "C" void kernel_launch(void* const* d_in, const int* in_sizes, int n_in,
                              void* d_out, int out_size, void* d_ws, size_t ws_size,
                              hipStream_t stream) {
  (void)in_sizes; (void)n_in; (void)out_size; (void)ws_size;
  const float* m     = (const float*)d_in[0];
  const int*   eidx  = (const int*)d_in[1];
  const float* Win   = (const float*)d_in[2];
  const float* bin   = (const float*)d_in[3];
  const float* Wout  = (const float*)d_in[4];
  const float* bout  = (const float*)d_in[5];
  float* z = (float*)d_out;

  // workspace layout
  unsigned*       wb = (unsigned*)d_ws;                              // 256 KB
  unsigned short* aB = (unsigned short*)((char*)d_ws + 262144);      // 3.2 MB
  unsigned short* bB = (unsigned short*)((char*)d_ws + 262144 + 3200000);

  prep_wb_kernel<<<256, 256, 0, stream>>>(Wout, wb);
  phase1_kernel<<<N_NODE / 16, 256, 0, stream>>>(m, Win, bin, aB, bB);

  const size_t shBytes = 262144 + 8 * 1024 * sizeof(unsigned);       // 294912
  pair_outer_kernel<<<512, 256, shBytes, stream>>>(eidx, aB, bB, wb, bout, z);
}
